// _FakeSparseMoeBlock_9302899163575
// MI455X (gfx1250) — matrix operand order repaired and checked by reading, not compiled, not run on hardware
//
#include <hip/hip_runtime.h>
#include <stdint.h>

#define HIDDEN 2048
#define INTER  1408
#define NEXP   8

typedef __attribute__((ext_vector_type(16))) __bf16 v16bf;
typedef __attribute__((ext_vector_type(8)))  float  v8f;

union Frag16 { v16bf v; uint4 q[2]; };
union FragC  { v8f v; float f[8]; };

// native f32->bf16 (RNE) -> packed dword; should lower to v_cvt_pk_bf16_f32
__device__ __forceinline__ uint32_t pack2bf(float a, float b) {
  union { __bf16 h[2]; uint32_t u; } o;
  o.h[0] = (__bf16)a; o.h[1] = (__bf16)b;
  return o.u;
}

// async global->LDS copy, 16B per lane (CDNA5, ASYNCcnt-tracked)
__device__ __forceinline__ void async_load_b128(uint32_t lds_off, const void* gaddr) {
  asm volatile("global_load_async_to_lds_b128 %0, %1, off"
               :: "v"(lds_off), "v"(gaddr) : "memory");
}
__device__ __forceinline__ void wait_async() {
  asm volatile("s_wait_asynccnt 0x0" ::: "memory");
}

// ---------------- Kernel 1: router (logits -> softmax -> top2 -> renorm) ----
__global__ __launch_bounds__(128) void moe_router(
    const float* __restrict__ x, const float* __restrict__ rw,
    int* __restrict__ tope, float* __restrict__ topw, int T) {
  int wave = threadIdx.x >> 5;
  int lane = threadIdx.x & 31;
  int t = blockIdx.x * 4 + wave;
  if (t >= T) return;
  const float* xr = x + (size_t)t * HIDDEN;
  float acc[NEXP];
#pragma unroll
  for (int e = 0; e < NEXP; ++e) acc[e] = 0.f;
  for (int d = lane; d < HIDDEN; d += 32) {
    float xv = xr[d];
#pragma unroll
    for (int e = 0; e < NEXP; ++e) acc[e] += xv * rw[e * HIDDEN + d];
  }
#pragma unroll
  for (int e = 0; e < NEXP; ++e)
#pragma unroll
    for (int o = 16; o > 0; o >>= 1) acc[e] += __shfl_xor(acc[e], o, 32);
  if (lane == 0) {
    float mx = acc[0];
#pragma unroll
    for (int e = 1; e < NEXP; ++e) mx = fmaxf(mx, acc[e]);
    float p[NEXP];
#pragma unroll
    for (int e = 0; e < NEXP; ++e) p[e] = expf(acc[e] - mx);
    int i0 = 0;
#pragma unroll
    for (int e = 1; e < NEXP; ++e) if (p[e] > p[i0]) i0 = e;
    int i1 = (i0 == 0) ? 1 : 0;
#pragma unroll
    for (int e = 0; e < NEXP; ++e) if (e != i0 && p[e] > p[i1]) i1 = e;
    float a = p[i0], b = p[i1];
    float inv = 1.f / (a + b);
    tope[t * 2 + 0] = i0; tope[t * 2 + 1] = i1;
    topw[t * 2 + 0] = a * inv; topw[t * 2 + 1] = b * inv;
  }
}

// ---------------- Kernel 2: deterministic compaction (wave per expert) ------
__global__ __launch_bounds__(256) void moe_compact(
    const int* __restrict__ tope, const float* __restrict__ topw,
    int* __restrict__ rows, float* __restrict__ wvals,
    int* __restrict__ counts, int* __restrict__ bases, int T) {
  int e = threadIdx.x >> 5;          // 8 waves, one per expert
  int lane = threadIdx.x & 31;
  __shared__ int scnt[NEXP];
  unsigned lmask = (1u << lane) - 1u;
  int cnt = 0;
  for (int base = 0; base < T; base += 32) {
    int t = base + lane;
    int e0 = tope[t * 2], e1 = tope[t * 2 + 1];
#pragma unroll
    for (int k = 0; k < 2; ++k) {
      bool sel = ((k == 0 ? e0 : e1) == e);
      unsigned m = (unsigned)__ballot(sel);
      if (sel) {
        int slot = cnt + __popc(m & lmask);
        rows[e * 4096 + slot] = t * 2 + k;
        wvals[e * 4096 + slot] = topw[t * 2 + k];
      }
      cnt += __popc(m);
    }
  }
  if (lane == 0) scnt[e] = cnt;
  __syncthreads();
  if (threadIdx.x == 0) {
    int run = 0;
    for (int i = 0; i < NEXP; ++i) { counts[i] = scnt[i]; bases[i] = run; run += scnt[i]; }
  }
}

#define LDA 40   // ushort row stride in LDS (16B-aligned fragment reads, bank spread)

// A fragment (16x32 bf16): M = lane&15 within wave tile, K = (lane&16)+j contiguous
__device__ __forceinline__ Frag16 load_fragA(const uint16_t* Xs, int mtw, int lane) {
  Frag16 a;
  int am = mtw + (lane & 15);
  int kb = (lane & 16) ? 16 : 0;
  a.q[0] = *(const uint4*)&Xs[am * LDA + kb];
  a.q[1] = *(const uint4*)&Xs[am * LDA + 8 + kb];
  return a;
}
// B fragment (32x16 bf16): N = lane&15 within tile, K = (lane&16)+j contiguous
__device__ __forceinline__ Frag16 load_fragB(const uint16_t* Ws, int ntile, int lane) {
  Frag16 b;
  const uint16_t* bp = &Ws[(ntile * 16 + (lane & 15)) * LDA + (lane & 16)];
  b.q[0] = *(const uint4*)bp;
  b.q[1] = *(const uint4*)(bp + 8);
  return b;
}

// ---------------- Kernel 3: gathered gate/up GEMM + SiLU -> h (bf16) --------
__global__ __launch_bounds__(256) void moe_gateup(
    const float* __restrict__ x, const float* __restrict__ gup,
    const int* __restrict__ rows, const int* __restrict__ counts,
    const int* __restrict__ bases, uint16_t* __restrict__ hbuf) {
  int e  = blockIdx.y >> 6;
  int mt = blockIdx.y & 63;
  int cnt = counts[e];
  int m0 = mt * 64;
  if (m0 >= cnt) return;
  int n0 = blockIdx.x * 64;          // inter-column base (22 blocks * 64 = 1408)

  __shared__ uint16_t Xs [2][64 * LDA];
  __shared__ uint16_t Wgs[2][64 * LDA];
  __shared__ uint16_t Wus[2][64 * LDA];
  __shared__ float Hg[64][65];
  __shared__ float Hu[64][65];
  __shared__ int rowid[64];

  int tid = threadIdx.x;
  if (tid < 64) {
    int s = m0 + tid;
    rowid[tid] = (s < cnt) ? rows[e * 4096 + s] : rows[e * 4096];
  }
  __syncthreads();

  const float* wg = gup + (size_t)e * (2 * INTER) * HIDDEN;
  int lm = tid >> 2;                 // staging row 0..63
  int ks = (tid & 3) * 8;            // staging k offset
  int wave = tid >> 5, lane = tid & 31;
  int matsel = wave >> 2;            // 0 = gate, 1 = up
  int mtw = (wave & 3) * 16;

  const float* xrow  = x  + (size_t)(rowid[lm] >> 1) * HIDDEN + ks;
  const float* wgrow = wg + (size_t)(n0 + lm) * HIDDEN + ks;
  const float* wurow = wg + (size_t)(INTER + n0 + lm) * HIDDEN + ks;

  auto stage = [&](int buf, int kk) {
    {
      float4 f0 = *(const float4*)(xrow + kk);
      float4 f1 = *(const float4*)(xrow + kk + 4);
      uint4 o = { pack2bf(f0.x, f0.y), pack2bf(f0.z, f0.w),
                  pack2bf(f1.x, f1.y), pack2bf(f1.z, f1.w) };
      *(uint4*)&Xs[buf][lm * LDA + ks] = o;
    }
    {
      float4 f0 = *(const float4*)(wgrow + kk);
      float4 f1 = *(const float4*)(wgrow + kk + 4);
      uint4 o = { pack2bf(f0.x, f0.y), pack2bf(f0.z, f0.w),
                  pack2bf(f1.x, f1.y), pack2bf(f1.z, f1.w) };
      *(uint4*)&Wgs[buf][lm * LDA + ks] = o;
    }
    {
      float4 f0 = *(const float4*)(wurow + kk);
      float4 f1 = *(const float4*)(wurow + kk + 4);
      uint4 o = { pack2bf(f0.x, f0.y), pack2bf(f0.z, f0.w),
                  pack2bf(f1.x, f1.y), pack2bf(f1.z, f1.w) };
      *(uint4*)&Wus[buf][lm * LDA + ks] = o;
    }
  };

  FragC c[4];
#pragma unroll
  for (int i = 0; i < 4; ++i)
#pragma unroll
    for (int r = 0; r < 8; ++r) c[i].f[r] = 0.f;

  stage(0, 0);
  __syncthreads();
  int cur = 0;
  for (int kk = 0; kk < HIDDEN; kk += 32) {
    if (kk + 32 < HIDDEN) stage(cur ^ 1, kk + 32);   // prefetch next chunk
    Frag16 a = load_fragA(&Xs[cur][0], mtw, lane);
    const uint16_t* Wsel = matsel ? &Wus[cur][0] : &Wgs[cur][0];
#pragma unroll
    for (int nt = 0; nt < 4; ++nt) {
      Frag16 b = load_fragB(Wsel, nt, lane);
      c[nt].v = __builtin_amdgcn_wmma_f32_16x16x32_bf16(
          false, a.v, false, b.v, (short)0, c[nt].v, false, false);
    }
    __syncthreads();
    cur ^= 1;
  }

  // spill per-wave tiles to LDS so gate/up meet elementwise
  {
    float (*H)[65] = matsel ? Hu : Hg;
#pragma unroll
    for (int nt = 0; nt < 4; ++nt) {
      int n = nt * 16 + (lane & 15);
#pragma unroll
      for (int r = 0; r < 8; ++r) {
        int m = mtw + r + ((lane >> 4) << 3);
        H[m][n] = c[nt].f[r];
      }
    }
  }
  __syncthreads();

  int hb = bases[e];
  for (int i = tid; i < 64 * 32; i += 256) {   // 2048 bf16 pairs
    int m  = i >> 5;
    int np = (i & 31) * 2;
    if (m0 + m < cnt) {
      float g0 = Hg[m][np], g1 = Hg[m][np + 1];
      float u0 = Hu[m][np], u1 = Hu[m][np + 1];
      float h0 = g0 / (1.f + expf(-g0)) * u0;
      float h1 = g1 / (1.f + expf(-g1)) * u1;
      *(uint32_t*)&hbuf[(size_t)(hb + m0 + m) * INTER + n0 + np] = pack2bf(h0, h1);
    }
  }
}

// ---------------- Kernel 4: down GEMM, scale by routing weight --------------
__global__ __launch_bounds__(256) void moe_down(
    const uint16_t* __restrict__ hbuf, const float* __restrict__ down,
    const int* __restrict__ rows, const float* __restrict__ wvals,
    const int* __restrict__ counts, const int* __restrict__ bases,
    float* __restrict__ part) {
  int e  = blockIdx.y >> 6;
  int mt = blockIdx.y & 63;
  int cnt = counts[e];
  int m0 = mt * 64;
  if (m0 >= cnt) return;
  int n0 = blockIdx.x * 64;          // output-column base (32 blocks * 64 = 2048)

  __shared__ uint16_t As[2][64 * LDA];
  __shared__ uint16_t Ws[2][64 * LDA];
  __shared__ int   rowid[64];
  __shared__ float wrow[64];

  int tid = threadIdx.x;
  if (tid < 64) {
    int s = m0 + tid;
    bool v = s < cnt;
    rowid[tid] = v ? rows[e * 4096 + s] : 0;
    wrow[tid]  = v ? wvals[e * 4096 + s] : 0.f;
  }
  __syncthreads();

  int lm = tid >> 2, ks = (tid & 3) * 8;
  int wave = tid >> 5, lane = tid & 31;
  int mtw = (wave & 3) * 16;
  int ntb = (wave >> 2) * 2;

  const float* wd = down + (size_t)e * HIDDEN * INTER;
  const float*    wrowp = wd   + (size_t)(n0 + lm) * INTER + ks;
  const uint16_t* hrowp = hbuf + (size_t)(bases[e] + m0 + lm) * INTER + ks;

  auto stageA = [&](int buf, int kk) {   // h tile is already bf16 -> async copy
    async_load_b128((uint32_t)(uintptr_t)&As[buf][lm * LDA + ks], hrowp + kk);
  };
  auto stageW = [&](int buf, int kk) {
    float4 f0 = *(const float4*)(wrowp + kk);
    float4 f1 = *(const float4*)(wrowp + kk + 4);
    uint4 o = { pack2bf(f0.x, f0.y), pack2bf(f0.z, f0.w),
                pack2bf(f1.x, f1.y), pack2bf(f1.z, f1.w) };
    *(uint4*)&Ws[buf][lm * LDA + ks] = o;
  };

  FragC c[2];
#pragma unroll
  for (int i = 0; i < 2; ++i)
#pragma unroll
    for (int r = 0; r < 8; ++r) c[i].f[r] = 0.f;

  stageA(0, 0); stageW(0, 0);
  wait_async();
  __syncthreads();
  int cur = 0;
  for (int kk = 0; kk < INTER; kk += 32) {
    if (kk + 32 < INTER) { stageA(cur ^ 1, kk + 32); stageW(cur ^ 1, kk + 32); }
    Frag16 a = load_fragA(&As[cur][0], mtw, lane);
#pragma unroll
    for (int nt = 0; nt < 2; ++nt) {
      Frag16 b = load_fragB(&Ws[cur][0], ntb + nt, lane);
      c[nt].v = __builtin_amdgcn_wmma_f32_16x16x32_bf16(
          false, a.v, false, b.v, (short)0, c[nt].v, false, false);
    }
    wait_async();
    __syncthreads();
    cur ^= 1;
  }

#pragma unroll
  for (int nt = 0; nt < 2; ++nt) {
    int n = n0 + (ntb + nt) * 16 + (lane & 15);
#pragma unroll
    for (int r = 0; r < 8; ++r) {
      int m = mtw + r + ((lane >> 4) << 3);
      if (m0 + m < cnt) {
        int t2k = rowid[m];           // token*2 + k  (unique per assignment)
        part[(size_t)t2k * HIDDEN + n] = wrow[m] * c[nt].f[r];
      }
    }
  }
}

// ---------------- Kernel 5: combine the two assignments per token -----------
__global__ __launch_bounds__(256) void moe_combine(
    const float* __restrict__ part, float* __restrict__ out, int T) {
  size_t i = (size_t)blockIdx.x * blockDim.x + threadIdx.x;
  size_t n = (size_t)T * (HIDDEN / 4);
  if (i >= n) return;
  size_t t  = i / (HIDDEN / 4);
  size_t c4 = i % (HIDDEN / 4);
  float4 a = ((const float4*)(part + (t * 2 + 0) * (size_t)HIDDEN))[c4];
  float4 b = ((const float4*)(part + (t * 2 + 1) * (size_t)HIDDEN))[c4];
  float4 o = { a.x + b.x, a.y + b.y, a.z + b.z, a.w + b.w };
  ((float4*)(out + t * (size_t)HIDDEN))[c4] = o;
}

extern "C" void kernel_launch(void* const* d_in, const int* in_sizes, int n_in,
                              void* d_out, int out_size, void* d_ws, size_t ws_size,
                              hipStream_t stream) {
  const float* x   = (const float*)d_in[0];
  const float* rw  = (const float*)d_in[1];
  const float* gup = (const float*)d_in[2];
  const float* dwn = (const float*)d_in[3];
  float* out = (float*)d_out;
  int T = in_sizes[0] / HIDDEN;     // 4096

  char* ws = (char*)d_ws;
  size_t o = 0;
  int*      tope  = (int*)(ws + o);      o += (size_t)T * 2 * sizeof(int);
  float*    topw  = (float*)(ws + o);    o += (size_t)T * 2 * sizeof(float);
  int*      rows  = (int*)(ws + o);      o += (size_t)NEXP * 4096 * sizeof(int);
  float*    wvals = (float*)(ws + o);    o += (size_t)NEXP * 4096 * sizeof(float);
  int*      counts= (int*)(ws + o);      o += 256;
  int*      bases = (int*)(ws + o);      o += 256;
  uint16_t* hbuf  = (uint16_t*)(ws + o); o += (size_t)(2 * T + 64) * INTER * sizeof(uint16_t);
  o = (o + 255) & ~(size_t)255;
  float*    part  = (float*)(ws + o);    o += (size_t)(2 * T) * HIDDEN * sizeof(float);

  moe_router <<<(T + 3) / 4, 128, 0, stream>>>(x, rw, tope, topw, T);
  moe_compact<<<1, 256, 0, stream>>>(tope, topw, rows, wvals, counts, bases, T);
  moe_gateup <<<dim3(INTER / 64, NEXP * 64), 256, 0, stream>>>(x, gup, rows, counts, bases, hbuf);
  moe_down   <<<dim3(HIDDEN / 64, NEXP * 64), 256, 0, stream>>>(hbuf, dwn, rows, wvals, counts, bases, part);
  moe_combine<<<(T * (HIDDEN / 4) + 255) / 256, 256, 0, stream>>>(part, out, T);
}
